// GateModel_35588099014919
// MI455X (gfx1250) — matrix operand order repaired and checked by reading, not compiled, not run on hardware
//
#include <hip/hip_runtime.h>
#include <hip/hip_bf16.h>
#include <stddef.h>

typedef __attribute__((ext_vector_type(16))) __bf16 v16bf;
typedef __attribute__((ext_vector_type(8)))  float  v8f;

#define B_   64
#define L_   2048
#define H_   128
#define FF_  256
#define V_   32000
#define XS_S 132   // padded f32 stride: bank = (4m + j) mod 64 -> conflict-free over m
#define YS_S 264   // padded bf16 stride

// K index held by packed element e of the 16-bit A-matrix fragment:
// lanes 0-15: K 0..15 ; lanes 16-31: K 16..31 (matches contiguous B loads)
static __device__ __forceinline__ int a_kidx(int e, int lane){
  int v = e >> 1, p = e & 1;
  return (v << 1) + p + ((lane & 16) ? 16 : 0);
}

static __device__ __forceinline__ v8f zero8(){
  v8f z;
#pragma unroll
  for (int i = 0; i < 8; ++i) z[i] = 0.f;
  return z;
}

static __device__ __forceinline__ v8f wmma_bf16(v16bf a, v16bf b, v8f c){
  return __builtin_amdgcn_wmma_f32_16x16x32_bf16(false, a, false, b, (short)0, c,
                                                 false, false);
}

// 32B vector load of a pre-converted bf16 B-fragment (addresses are 32B aligned)
static __device__ __forceinline__ v16bf load16bf(const __bf16* p){
  return *(const v16bf*)p;
}

// =====================================================================
// Kernel W: one-time f32 -> bf16 weight conversion (grid-stride)
// =====================================================================
__global__ __launch_bounds__(256) void k_cvt(const float* __restrict__ src,
                                             __bf16* __restrict__ dst, int n)
{
  int i = blockIdx.x * 256 + threadIdx.x;
  const int stride = gridDim.x * 256;
  for (; i < n; i += stride) dst[i] = (__bf16)src[i];
}

// =====================================================================
// Kernel A: embed gather + FF(relu) + residual + LayerNorm + k-projection
// 1 wave handles a 16-token tile; 2 waves / block; grid = B*L/32
// =====================================================================
__global__ __launch_bounds__(64) void k_token(
    const int*    __restrict__ seq,  const float*  __restrict__ embW,
    const __bf16* __restrict__ w1b,  const float*  __restrict__ b1,
    const __bf16* __restrict__ w2b,  const float*  __restrict__ b2,
    const float*  __restrict__ lng,  const float*  __restrict__ lnb,
    const __bf16* __restrict__ kpwb, float* __restrict__ k_all)
{
  __shared__ float  xs_all[2][16 * XS_S];   // f32 token tile (h -> x -> hn)
  __shared__ __bf16 ys_all[2][16 * YS_S];   // bf16 post-relu FF activations
  const int wv   = threadIdx.x >> 5;
  const int lane = threadIdx.x & 31;
  float*  xs = xs_all[wv];
  __bf16* ys = ys_all[wv];
  const int tile = blockIdx.x * 2 + wv;
  const int tok0 = tile * 16;

  // ---- stage embeddings: xs[m][j] = embW[seq[tok0+m]][j]
  for (int i = lane; i < 16 * H_; i += 32){
    int m = i >> 7, j = i & 127;
    xs[m * XS_S + j] = embW[(size_t)seq[tok0 + m] * H_ + j];
  }
  __syncthreads();

  const int mloc = lane & 15;
  const int k16  = (lane & 16) ? 16 : 0;   // B-fragment K offset for high half-wave
  const int mb   = (lane & 16) ? 8  : 0;   // C/D row offset for high half-wave

  // ---- GEMM1: y = relu(h @ w1^T + b1)  (16x128)x(128x256)
  v16bf a1[4];
#pragma unroll
  for (int kt = 0; kt < 4; ++kt)
#pragma unroll
    for (int e = 0; e < 16; ++e)
      a1[kt][e] = (__bf16)xs[mloc * XS_S + kt * 32 + a_kidx(e, lane)];

#pragma unroll
  for (int nt = 0; nt < 16; ++nt){
    v8f acc = zero8();
    const int n0 = nt * 16 + mloc;
    const __bf16* wr = w1b + (size_t)n0 * H_ + k16;
#pragma unroll
    for (int kt = 0; kt < 4; ++kt)
      acc = wmma_bf16(a1[kt], load16bf(wr + kt * 32), acc);
    const float bias = b1[n0];
#pragma unroll
    for (int r = 0; r < 8; ++r)
      ys[(mb + r) * YS_S + n0] = (__bf16)fmaxf(acc[r] + bias, 0.f);
  }
  __syncthreads();

  // ---- GEMM2: ff = y @ w2^T + b2 ;  x = h + ff (in place in xs)
  v16bf a2[8];
#pragma unroll
  for (int kt = 0; kt < 8; ++kt)
#pragma unroll
    for (int e = 0; e < 16; ++e)
      a2[kt][e] = ys[mloc * YS_S + kt * 32 + a_kidx(e, lane)];

#pragma unroll
  for (int nt = 0; nt < 8; ++nt){
    v8f acc = zero8();
    const int n0 = nt * 16 + mloc;
    const __bf16* wr = w2b + (size_t)n0 * FF_ + k16;
#pragma unroll
    for (int kt = 0; kt < 8; ++kt)
      acc = wmma_bf16(a2[kt], load16bf(wr + kt * 32), acc);
    const float bias = b2[n0];
#pragma unroll
    for (int r = 0; r < 8; ++r){
      int idx = (mb + r) * XS_S + n0;
      xs[idx] = xs[idx] + acc[r] + bias;
    }
  }
  __syncthreads();

  // ---- LayerNorm per token (f32): lane pairs split the 128 columns
  {
    const int t = lane & 15, half = lane >> 4;
    float s = 0.f, s2 = 0.f;
#pragma unroll
    for (int j = 0; j < 64; ++j){
      float v = xs[t * XS_S + half * 64 + j];
      s += v; s2 += v * v;
    }
    s  += __shfl_xor(s,  16);
    s2 += __shfl_xor(s2, 16);
    float mu   = s  * (1.f / 128.f);
    float var  = s2 * (1.f / 128.f) - mu * mu;
    float rstd = rsqrtf(var + 1e-5f);
#pragma unroll
    for (int j = 0; j < 64; ++j){
      int c = half * 64 + j;
      float v = xs[t * XS_S + c];
      xs[t * XS_S + c] = (v - mu) * rstd * lng[c] + lnb[c];
    }
  }
  __syncthreads();

  // ---- GEMM3: k = hn @ kpw^T  (16x128)x(128x128), store f32 k_all
  v16bf a3[4];
#pragma unroll
  for (int kt = 0; kt < 4; ++kt)
#pragma unroll
    for (int e = 0; e < 16; ++e)
      a3[kt][e] = (__bf16)xs[mloc * XS_S + kt * 32 + a_kidx(e, lane)];

#pragma unroll
  for (int nt = 0; nt < 8; ++nt){
    v8f acc = zero8();
    const int n0 = nt * 16 + mloc;
    const __bf16* wr = kpwb + (size_t)n0 * H_ + k16;
#pragma unroll
    for (int kt = 0; kt < 4; ++kt)
      acc = wmma_bf16(a3[kt], load16bf(wr + kt * 32), acc);
#pragma unroll
    for (int r = 0; r < 8; ++r)
      k_all[(size_t)(tok0 + mb + r) * H_ + n0] = acc[r];
  }
}

// =====================================================================
// Kernel B: sequential fast-weight scan. 1 block / batch, 128 threads.
// Thread i owns M-row i in registers. energy = |k-vp|^2 * |kn|^2 / 16384.
// =====================================================================
__global__ __launch_bounds__(128) void k_scan(
    const float* __restrict__ k_all, const float* __restrict__ rp_w,
    const float* __restrict__ rp_b,  float* __restrict__ readp)
{
  const int b = blockIdx.x, tid = threadIdx.x;
  const int lane = tid & 31, wv = tid >> 5;
  __shared__ __align__(16) float kn_s[H_];
  __shared__ __align__(16) float rd_s[H_];
  __shared__ float red_s[8];

  float Mrow[H_];
#pragma unroll
  for (int j = 0; j < H_; ++j) Mrow[j] = 0.f;

  const float* kb = k_all + (size_t)b * L_ * H_;
  const float4* kn4 = (const float4*)kn_s;

#pragma unroll 1
  for (int t = 0; t < L_ - 1; ++t){
    float kt = kb[(size_t)t * H_ + tid];

    // reduce ||k||^2
    float ss = kt * kt;
#pragma unroll
    for (int d = 16; d; d >>= 1) ss += __shfl_xor(ss, d);
    if (lane == 0) red_s[wv] = ss;
    __syncthreads();
    ss = red_s[0] + red_s[1] + red_s[2] + red_s[3];
    float rn = 1.f / fmaxf(sqrtf(ss), 1e-12f);

    kn_s[tid] = kt * rn;
    __syncthreads();

    // vp_i = dot(Mrow_i, kn)
    float vp = 0.f;
#pragma unroll
    for (int j = 0; j < 32; ++j){
      float4 v = kn4[j];
      vp += Mrow[4*j]*v.x + Mrow[4*j+1]*v.y + Mrow[4*j+2]*v.z + Mrow[4*j+3]*v.w;
    }
    float e = kt - vp;

    // reduce ||k - vp||^2
    float E2 = e * e;
#pragma unroll
    for (int d = 16; d; d >>= 1) E2 += __shfl_xor(E2, d);
    if (lane == 0) red_s[4 + wv] = E2;
    __syncthreads();
    E2 = red_s[4] + red_s[5] + red_s[6] + red_s[7];

    float energy = E2 * (ss * rn * rn) * (1.f / 16384.f);
    if (energy >= 0.4f){             // uniform branch across block
#pragma unroll
      for (int j = 0; j < 32; ++j){
        float4 v = kn4[j];
        Mrow[4*j]   += e * v.x;
        Mrow[4*j+1] += e * v.y;
        Mrow[4*j+2] += e * v.z;
        Mrow[4*j+3] += e * v.w;
      }
    }
  }

  // q = k_all[:, L-1, :];  read = M @ q
  __syncthreads();
  kn_s[tid] = kb[(size_t)(L_ - 1) * H_ + tid];
  __syncthreads();
  float rdv = 0.f;
#pragma unroll
  for (int j = 0; j < 32; ++j){
    float4 v = kn4[j];
    rdv += Mrow[4*j]*v.x + Mrow[4*j+1]*v.y + Mrow[4*j+2]*v.z + Mrow[4*j+3]*v.w;
  }
  rd_s[tid] = rdv;
  __syncthreads();

  // readp_i = rp_b[i] + sum_j rp_w[i][j] * read[j]
  float acc = rp_b[tid];
  const float4* rw  = (const float4*)(rp_w + (size_t)tid * H_);
  const float4* rd4 = (const float4*)rd_s;
#pragma unroll
  for (int j = 0; j < 32; ++j){
    float4 w = rw[j]; float4 r = rd4[j];
    acc += w.x*r.x + w.y*r.y + w.z*r.z + w.w*r.w;
  }
  readp[b * H_ + tid] = acc;
}

// =====================================================================
// Kernel C: logits = readp @ out_w^T + out_b  (64x128)x(128x32000)
// 1 wave per 16x16 output tile; 8 waves / block; grid = 2000*4/8
// =====================================================================
__global__ __launch_bounds__(256) void k_out(
    const float*  __restrict__ readp, const __bf16* __restrict__ outwb,
    const float*  __restrict__ out_b, float* __restrict__ out)
{
  const int wid  = blockIdx.x * 8 + (threadIdx.x >> 5);
  const int lane = threadIdx.x & 31;
  const int mt = wid & 3;          // 0..3  (64 rows / 16)
  const int nt = wid >> 2;         // 0..1999
  const int mloc = lane & 15;
  const int k16  = (lane & 16) ? 16 : 0;

  // A fragments from readp (f32 -> bf16)
  const int mrow = mt * 16 + mloc;
  v16bf a[4];
#pragma unroll
  for (int kt = 0; kt < 4; ++kt){
    const float* ap = readp + (size_t)mrow * H_ + kt * 32;
#pragma unroll
    for (int e = 0; e < 16; ++e) a[kt][e] = (__bf16)ap[a_kidx(e, lane)];
  }

  const int n0 = nt * 16 + mloc;
  const __bf16* wp = outwb + (size_t)n0 * H_ + k16;
  __builtin_prefetch(wp + H_, 0, 1);     // next weight row -> global_prefetch

  v8f acc = zero8();
#pragma unroll
  for (int kt = 0; kt < 4; ++kt)
    acc = wmma_bf16(a[kt], load16bf(wp + kt * 32), acc);

  const float ob = out_b[n0];
  const int mbase = mt * 16 + ((lane & 16) ? 8 : 0);
#pragma unroll
  for (int r = 0; r < 8; ++r)
    out[(size_t)(mbase + r) * V_ + n0] = acc[r] + ob;
}

// =====================================================================
extern "C" void kernel_launch(void* const* d_in, const int* in_sizes, int n_in,
                              void* d_out, int out_size, void* d_ws, size_t ws_size,
                              hipStream_t stream)
{
  (void)in_sizes; (void)n_in; (void)out_size; (void)ws_size;
  const int*   seq  = (const int*)  d_in[0];
  const float* embW = (const float*)d_in[1];
  const float* w1   = (const float*)d_in[2];
  const float* b1   = (const float*)d_in[3];
  const float* w2   = (const float*)d_in[4];
  const float* b2   = (const float*)d_in[5];
  const float* lng  = (const float*)d_in[6];
  const float* lnb  = (const float*)d_in[7];
  const float* kpw  = (const float*)d_in[8];
  const float* rpw  = (const float*)d_in[9];
  const float* rpb  = (const float*)d_in[10];
  const float* outw = (const float*)d_in[11];
  const float* outb = (const float*)d_in[12];

  // workspace layout
  char* ws = (char*)d_ws;
  float*  k_all = (float*)ws;                              ws += (size_t)B_ * L_ * H_ * 4;  // 64 MB
  float*  readp = (float*)ws;                              ws += (size_t)B_ * H_ * 4;       // 32 KB
  __bf16* w1b   = (__bf16*)ws;                             ws += (size_t)FF_ * H_ * 2;      // 64 KB
  __bf16* w2b   = (__bf16*)ws;                             ws += (size_t)H_ * FF_ * 2;      // 64 KB
  __bf16* kpwb  = (__bf16*)ws;                             ws += (size_t)H_ * H_ * 2;       // 32 KB
  __bf16* outwb = (__bf16*)ws;                                                              // 8 MB

  // one-time weight conversion to bf16
  hipLaunchKernelGGL(k_cvt, dim3(64),   dim3(256), 0, stream, w1,   w1b,  FF_ * H_);
  hipLaunchKernelGGL(k_cvt, dim3(64),   dim3(256), 0, stream, w2,   w2b,  H_ * FF_);
  hipLaunchKernelGGL(k_cvt, dim3(32),   dim3(256), 0, stream, kpw,  kpwb, H_ * H_);
  hipLaunchKernelGGL(k_cvt, dim3(2048), dim3(256), 0, stream, outw, outwb, V_ * H_);

  hipLaunchKernelGGL(k_token, dim3((B_ * L_) / 32), dim3(64), 0, stream,
                     seq, embW, w1b, b1, w2b, b2, lng, lnb, kpwb, k_all);
  hipLaunchKernelGGL(k_scan, dim3(B_), dim3(128), 0, stream,
                     k_all, rpw, rpb, readp);
  hipLaunchKernelGGL(k_out, dim3((V_ / 16) * (B_ / 16) / 8), dim3(256), 0, stream,
                     readp, outwb, outb, (float*)d_out);
}
